// Encoder_72284299592169
// MI455X (gfx1250) — matrix operand order repaired and checked by reading, not compiled, not run on hardware
//
#include <hip/hip_runtime.h>
#include <hip/hip_bf16.h>
#include <math.h>

// ---------------- problem constants (from reference) ----------------
#define KP   64      // particles
#define BB   128     // batch
#define TT   48      // timesteps
#define INW  64      // input size
#define HH   256     // hidden
#define MROW (KP*BB) // 8192 particle rows
#define KD   (INW+HH)      // 320 packed GEMM K
#define GN   (4*HH)        // 1024 gate columns

typedef __attribute__((ext_vector_type(16))) _Float16 v16h;
typedef __attribute__((ext_vector_type(8)))  _Float16 v8h;
typedef __attribute__((ext_vector_type(8)))  float    v8f;
typedef __attribute__((ext_vector_type(4)))  unsigned int v4u;
typedef __attribute__((ext_vector_type(8)))  int v8i;
typedef __attribute__((ext_vector_type(4)))  int v4i;

// ---------------- deterministic hash RNG ----------------
__device__ __forceinline__ unsigned mixu(unsigned x){
  x ^= x >> 16; x *= 0x7FEB352Du; x ^= x >> 15; x *= 0x846CA68Bu; x ^= x >> 16; return x;
}
__device__ __forceinline__ float u01(unsigned h){
  return ((h >> 8) + 0.5f) * (1.0f / 16777216.0f);
}
__device__ __forceinline__ float nrand(int t, int r, int h){
  unsigned s  = mixu(0x243F6A88u ^ (unsigned)t * 0x9E3779B1u ^ (unsigned)r * 0x85EBCA77u
                                 ^ (unsigned)h * 0xC2B2AE3Du);
  unsigned s2 = mixu(s ^ 0x13198A2Eu);
  float rr = sqrtf(-2.0f * __logf(u01(s)));
  return rr * __cosf(6.28318530718f * u01(s2));
}
__device__ __forceinline__ float sigf(float x){ return 1.0f / (1.0f + __expf(-x)); }

// ---------------- utility kernels ----------------
__global__ void k_zero(float* p, int n){
  for (int i = blockIdx.x * blockDim.x + threadIdx.x; i < n; i += gridDim.x * blockDim.x)
    p[i] = 0.0f;
}

// pack [W_ih | W_hh] rows to f16, bias sum b_ih+b_hh
__global__ void k_pack_w(const float* __restrict__ Wih, const float* __restrict__ Whh,
                         const float* __restrict__ bih, const float* __restrict__ bhh,
                         _Float16* __restrict__ Wcat, float* __restrict__ bcat){
  int gid = blockIdx.x * blockDim.x + threadIdx.x;
  if (gid >= GN * KD) return;
  int n = gid / KD, c = gid % KD;
  Wcat[gid] = (c < INW) ? (_Float16)Wih[n * INW + c] : (_Float16)Whh[n * HH + (c - INW)];
  if (c == 0) bcat[n] = bih[n] + bhh[n];
}

// mean over particles: (K*B,H) -> (B,H)
__global__ void k_state_mean(const float* __restrict__ hid, const float* __restrict__ cel,
                             float* __restrict__ hm, float* __restrict__ cm){
  int gid = blockIdx.x * blockDim.x + threadIdx.x;   // B*H
  if (gid >= BB * HH) return;
  float sh = 0.f, sc = 0.f;
  for (int k = 0; k < KP; ++k){
    size_t off = (size_t)k * BB * HH + gid;
    sh += hid[off]; sc += cel[off];
  }
  hm[gid] = sh * (1.0f / KP); cm[gid] = sc * (1.0f / KP);
}

// input attention + softmax over IN; writes new_input and input_weighted[:,t,:]
__global__ void k_attn(const float* __restrict__ inp, const float* __restrict__ Wa,
                       const float* __restrict__ ba, const float* __restrict__ hm,
                       const float* __restrict__ cm, float* __restrict__ newin,
                       float* __restrict__ out_iw, int t){
  __shared__ float red[INW];
  __shared__ float s0s;
  int j = threadIdx.x, b = blockIdx.x;
  float partial = 0.f;
  for (int c = j; c < 2 * HH; c += INW){
    float xv = (c < HH) ? hm[b * HH + c] : cm[b * HH + (c - HH)];
    partial += Wa[c] * xv;
  }
  red[j] = partial;
  for (int off = INW / 2; off > 0; off >>= 1){ __syncthreads(); if (j < off) red[j] += red[j + off]; }
  __syncthreads();
  if (j == 0) s0s = red[0] + ba[0];
  __syncthreads();
  float sc = s0s;
  const float* xb = inp + (size_t)b * TT * INW;
  for (int s = 0; s < TT; ++s) sc += Wa[2 * HH + s] * xb[s * INW + j];
  __syncthreads();
  red[j] = sc;
  for (int off = INW / 2; off > 0; off >>= 1){ __syncthreads(); if (j < off) red[j] = fmaxf(red[j], red[j + off]); }
  __syncthreads();
  float mx = red[0];
  __syncthreads();
  float e = __expf(sc - mx);
  red[j] = e;
  for (int off = INW / 2; off > 0; off >>= 1){ __syncthreads(); if (j < off) red[j] += red[j + off]; }
  __syncthreads();
  float ni = (e / red[0]) * xb[t * INW + j];
  newin[b * INW + j] = ni;
  out_iw[(size_t)b * TT * INW + (size_t)t * INW + j] = ni;
}

// std = softplus( [new_input, hmean] @ W_var^T + b_var )
__global__ void k_var_std(const float* __restrict__ newin, const float* __restrict__ hm,
                          const float* __restrict__ Wv, const float* __restrict__ bv,
                          float* __restrict__ stdbuf){
  int gid = blockIdx.x * blockDim.x + threadIdx.x;   // B*H
  if (gid >= BB * HH) return;
  int b = gid >> 8, h = gid & (HH - 1);
  const float* wr = Wv + (size_t)h * KD;
  float s = bv[h];
  for (int c = 0; c < INW; ++c) s += newin[b * INW + c] * wr[c];
  for (int c = 0; c < HH; ++c)  s += hm[b * HH + c] * wr[INW + c];
  stdbuf[gid] = (s > 20.f) ? s : log1pf(__expf(s));
}

// Build f16 activation matrix A = [x_l | hiddens] (8192 x 320)
__global__ void k_pack_a(const float* __restrict__ newin, const float* __restrict__ hid,
                         _Float16* __restrict__ A){
  int gid = blockIdx.x * blockDim.x + threadIdx.x;
  if (gid >= MROW * KD) return;
  int r = gid / KD, c = gid % KD;
  A[gid] = (c < INW) ? (_Float16)newin[(r & (BB - 1)) * INW + c]
                     : (_Float16)hid[(size_t)r * HH + (c - INW)];
}

// ---------------- TDM weight stage: 3D tile (320 f16 x 16 rows x 4 gate groups) ----------------
// D# per CDNA5 ISA: group0 = {flags, lds_addr, global_addr, type},
// group1 = dims/strides, group2 = dim2, group3 = unused.
// amdgpu-toolchain (clang-23) builtin arity: (v4u, v8i, v4i, v4i, v8i, i32 cpol).
__device__ __forceinline__ void tdm_load_weights(const _Float16* gsrc, _Float16* lds_dst){
  unsigned lds_addr = (unsigned)(size_t)lds_dst;
  unsigned long long ga = (unsigned long long)(size_t)gsrc;
  v4u g0;
  g0[0] = 1u;                                                // count=1, is_restore=0
  g0[1] = lds_addr;                                          // lds_addr       [63:32]
  g0[2] = (unsigned)(ga & 0xFFFFFFFFu);                      // global_addr    [95:64]
  g0[3] = (unsigned)((ga >> 32) & 0x01FFFFFFu) | (2u << 30); // ga[56:32] + type=2 [127:126]
  const unsigned td0 = KD;            // tensor_dim0 (f16 units)
  const unsigned td1 = GN;            // tensor_dim1 = 1024 rows
  const unsigned t0s = KD;            // tensor_dim0_stride (48-bit; high bits 0)
  const unsigned t1s = HH * KD;       // tensor_dim1_stride = 256 rows between gate groups
  v8i g1;
  g1[0] = (int)(1u << 16);                                   // data_size=1 (2B), wg_mask=0
  g1[1] = (int)((td0 & 0xFFFFu) << 16);                      // tensor_dim0 lo16 [63:48]
  g1[2] = (int)(((td0 >> 16) & 0xFFFFu) | ((td1 & 0xFFFFu) << 16)); // dim0 hi / dim1 lo
  g1[3] = (int)(((td1 >> 16) & 0xFFFFu) | ((unsigned)KD << 16));    // dim1 hi / tile_dim0=320
  g1[4] = (int)(16u | (4u << 16));                           // tile_dim1=16, tile_dim2=4
  g1[5] = (int)t0s;                                          // tensor_dim0_stride [191:160]
  g1[6] = (int)((t1s & 0xFFFFu) << 16);                      // d0s hi16 (=0) / d1s lo16
  g1[7] = (int)((t1s >> 16) & 0xFFFFFFFFu);                  // tensor_dim1_stride hi32
  v4i g2 = { (int)4, 0, 0, 0 };                              // tensor_dim2 = 4
  v4i g3 = { 0, 0, 0, 0 };
  v8i g4 = { 0, 0, 0, 0, 0, 0, 0, 0 };                       // unused group (zeroed)
  __builtin_amdgcn_tensor_load_to_lds(g0, g1, g2, g3, g4, 0);
}

// ---------------- fused WMMA LSTM step ----------------
// One block per N-tile x M-group: TDM stages this N-tile's 64 weight rows (40KB)
// into LDS once; the compiler promotes the 40 B-fragments to registers
// (extended-VGPR file via s_set_vgpr_msb) and each of 8 waves sweeps 8 M-tiles.
// The A-fragment is explicitly software-pipelined so chunk k+1's loads are in
// flight while chunk k's 4 WMMAs execute.
__global__ __launch_bounds__(256) void k_lstm_wmma(
    const _Float16* __restrict__ A, const _Float16* __restrict__ W,
    const float* __restrict__ bcat, const float* __restrict__ cells,
    const float* __restrict__ stdbuf, float* __restrict__ hid_tmp,
    float* __restrict__ cells_tmp, int t){
  __shared__ __align__(64) _Float16 Wlds[64 * KD];           // 40 KB
  int wave = threadIdx.x >> 5;
  int lane = threadIdx.x & 31;
  int ntile  = blockIdx.x & 15;            // hidden-column tile
  int mgroup = blockIdx.x >> 4;            // 8 groups of 64 M-tiles
  int lrow  = lane & 15;
  int lhalf = lane >> 4;

  if (wave == 0){
    tdm_load_weights(W + (size_t)(ntile * 16) * KD, Wlds);   // rows n = g*256 + ntile*16 + y
    __builtin_amdgcn_s_wait_tensorcnt(0);
  }
  __syncthreads();

  int h = ntile * 16 + lrow;
  float bi = bcat[h], bf = bcat[HH + h], bg = bcat[2 * HH + h], bo = bcat[3 * HH + h];

#pragma unroll 1
  for (int mi = 0; mi < 8; ++mi){
    int mtile = mgroup * 64 + mi * 8 + wave;
    const _Float16* abase = A + (size_t)(mtile * 16 + lrow) * KD;
    // prefetch next M-tile's A row (640B) while this tile computes
    if (mi + 1 < 8){
      const _Float16* anext = A + (size_t)((mtile + 8) * 16 + lrow) * KD;
      __builtin_prefetch(anext, 0, 1);                       // global_prefetch_b8
      __builtin_prefetch(anext + 256, 0, 1);
    }
    v8f acc0 = {}, acc1 = {}, acc2 = {}, acc3 = {};

    // software-pipelined A fragment (double buffer)
    v8h alo = *(const v8h*)(abase + lhalf * 16);
    v8h ahi = *(const v8h*)(abase + 8 + lhalf * 16);
#pragma unroll
    for (int kc = 0; kc < KD; kc += 32){
      v8h nlo = alo, nhi = ahi;
      if (kc + 32 < KD){
        nlo = *(const v8h*)(abase + kc + 32 + lhalf * 16);
        nhi = *(const v8h*)(abase + kc + 40 + lhalf * 16);
      }
      v16h a;
#pragma unroll
      for (int i = 0; i < 8; ++i){ a[i] = alo[i]; a[i + 8] = ahi[i]; }
#pragma unroll
      for (int g = 0; g < 4; ++g){
        const v16h bfrag = *(const v16h*)&Wlds[(g * 16 + lrow) * KD + kc + lhalf * 16];
        v8f acc = (g == 0) ? acc0 : (g == 1) ? acc1 : (g == 2) ? acc2 : acc3;
        acc = __builtin_amdgcn_wmma_f32_16x16x32_f16(false, a, false, bfrag,
                                                     (short)0, acc, false, false);
        if (g == 0) acc0 = acc; else if (g == 1) acc1 = acc;
        else if (g == 2) acc2 = acc; else acc3 = acc;
      }
      alo = nlo; ahi = nhi;
    }

    // Epilogue: C layout -> row = mtile*16 + v + 8*lhalf, col(h) = ntile*16 + (lane&15)
#pragma unroll
    for (int v = 0; v < 8; ++v){
      int row = mtile * 16 + v + 8 * lhalf;
      float gi = acc0[v] + bi;
      float gf = acc1[v] + bf;
      float gg = acc2[v] + bg;
      float go = acc3[v] + bo;
      float co = cells[(size_t)row * HH + h];
      float cn = sigf(gf) * co + sigf(gi) * tanhf(gg);
      float hmu = sigf(go) * tanhf(cn);
      int b = row & (BB - 1);
      float sd  = stdbuf[b * HH + h];
      float eps = nrand(t, row, h);
      hid_tmp[(size_t)row * HH + h]   = hmu + eps * sd;
      cells_tmp[(size_t)row * HH + h] = cn;
    }
  }
}

// scalar projection + bitonic argsort (ascending) per batch over K=64 particles
__global__ void k_fc_sort(const float* __restrict__ hid_tmp, const float* __restrict__ Wfc,
                          const float* __restrict__ bfc, int* __restrict__ sortidx){
  __shared__ float sv[KP];
  __shared__ int   si[KP];
  int k = threadIdx.x, b = blockIdx.x;
  const float* hr = hid_tmp + (size_t)(k * BB + b) * HH;
  float p = bfc[0];
  for (int c = 0; c < HH; ++c) p += Wfc[c] * hr[c];
  sv[k] = p; si[k] = k;
  __syncthreads();
  for (int size = 2; size <= KP; size <<= 1){
    for (int stride = size >> 1; stride > 0; stride >>= 1){
      int partner = k ^ stride;
      float myv = sv[k]; int myi = si[k];
      float ov  = sv[partner]; int oi = si[partner];
      __syncthreads();
      bool ascending   = ((k & size) == 0);
      bool keepSmaller = ((k < partner) == ascending);
      bool takeOther   = keepSmaller ? (ov < myv) : (ov > myv);
      if (takeOther){ sv[k] = ov; si[k] = oi; } else { sv[k] = myv; si[k] = myi; }
      __syncthreads();
    }
  }
  sortidx[b * KP + k] = si[k];
}

// hid_sorted[(k*B+b)] = hid_tmp[(sortidx[b][k]*B+b)]
__global__ void k_perm_sort(const float* __restrict__ hid_tmp, const int* __restrict__ sortidx,
                            float* __restrict__ hid_sorted){
  int gid = blockIdx.x * blockDim.x + threadIdx.x;   // MROW*HH
  if (gid >= MROW * HH) return;
  int r = gid >> 8, h = gid & (HH - 1);
  int k = r >> 7, b = r & (BB - 1);
  int src = sortidx[b * KP + k];
  hid_sorted[gid] = hid_tmp[((size_t)(src * BB + b)) * HH + h];
}

// particle weights: prob = exp([hid, x_l] @ W_pdf^T + b_pdf)
__global__ void k_pdf(const float* __restrict__ hid_sorted, const float* __restrict__ newin,
                      const float* __restrict__ Wp, const float* __restrict__ bp,
                      float* __restrict__ prob){
  int r = blockIdx.x * blockDim.x + threadIdx.x;     // MROW
  if (r >= MROW) return;
  int b = r & (BB - 1);
  const float* hr = hid_sorted + (size_t)r * HH;
  float s = bp[0];
  for (int c = 0; c < HH; ++c)  s += Wp[c] * hr[c];
  for (int c = 0; c < INW; ++c) s += Wp[HH + c] * newin[b * INW + c];
  prob[r] = __expf(s);
}

// categorical with replacement per batch (deterministic hash uniforms)
__global__ void k_resample(const float* __restrict__ prob, int t, int* __restrict__ idx2){
  __shared__ float w[KP];
  __shared__ float cs[KP];
  int k = threadIdx.x, b = blockIdx.x;
  w[k] = prob[k * BB + b];
  __syncthreads();
  if (k == 0){ float a = 0.f; for (int j = 0; j < KP; ++j){ a += w[j]; cs[j] = a; } }
  __syncthreads();
  float u = u01(mixu(0x9E3779B9u ^ (unsigned)t * 0x85EBCA77u ^ (unsigned)b * 0xC2B2AE3Du
                               ^ (unsigned)k * 0x27D4EB2Fu)) * cs[KP - 1];
  int pick = KP - 1;
  for (int j = 0; j < KP; ++j){ if (cs[j] >= u){ pick = j; break; } }
  idx2[b * KP + k] = pick;
}

// gather resampled hiddens (from sorted) and cells (from unsorted c_new)
__global__ void k_perm_resample(const float* __restrict__ hid_sorted,
                                const float* __restrict__ cells_tmp,
                                const int* __restrict__ idx2,
                                float* __restrict__ hiddens, float* __restrict__ cells){
  int gid = blockIdx.x * blockDim.x + threadIdx.x;   // MROW*HH
  if (gid >= MROW * HH) return;
  int r = gid >> 8, h = gid & (HH - 1);
  int k = r >> 7, b = r & (BB - 1);
  size_t src = ((size_t)(idx2[b * KP + k] * BB + b)) * HH + h;
  hiddens[gid] = hid_sorted[src];
  cells[gid]   = cells_tmp[src];
}

// encoder output: mean over all K*B rows, broadcast over batch into d_out
__global__ void k_enc_mean(const float* __restrict__ hiddens, float* __restrict__ out_enc, int t){
  __shared__ float red[256];
  int h = blockIdx.x, tid = threadIdx.x;
  float s = 0.f;
  for (int r = tid; r < MROW; r += 256) s += hiddens[(size_t)r * HH + h];
  red[tid] = s;
  for (int off = 128; off > 0; off >>= 1){ __syncthreads(); if (tid < off) red[tid] += red[tid + off]; }
  __syncthreads();
  float m = red[0] * (1.0f / MROW);
  if (tid < BB) out_enc[(size_t)tid * TT * HH + (size_t)t * HH + h] = m;
}

// ---------------- host-side launch ----------------
extern "C" void kernel_launch(void* const* d_in, const int* in_sizes, int n_in,
                              void* d_out, int out_size, void* d_ws, size_t ws_size,
                              hipStream_t stream) {
  (void)in_sizes; (void)n_in; (void)out_size; (void)ws_size;
  const float* input_data = (const float*)d_in[0];
  const float* W_attn = (const float*)d_in[1];
  const float* b_attn = (const float*)d_in[2];
  const float* W_var  = (const float*)d_in[3];
  const float* b_var  = (const float*)d_in[4];
  const float* W_ih   = (const float*)d_in[5];
  const float* b_ih   = (const float*)d_in[6];
  const float* W_hh   = (const float*)d_in[7];
  const float* b_hh   = (const float*)d_in[8];
  const float* W_fc   = (const float*)d_in[9];
  const float* b_fc   = (const float*)d_in[10];
  const float* W_pdf  = (const float*)d_in[11];
  const float* b_pdf  = (const float*)d_in[12];

  float* out_iw  = (float*)d_out;                         // (B,T,IN)
  float* out_enc = (float*)d_out + (size_t)BB * TT * INW; // (B,T,H)

  // carve workspace (256B aligned chunks)
  char* w = (char*)d_ws;
  auto carve = [&](size_t bytes) -> void* {
    void* p = (void*)w;
    w += (bytes + 255) & ~(size_t)255;
    return p;
  };
  const size_t stateB = (size_t)MROW * HH * sizeof(float);   // 8 MiB
  float*    hiddens    = (float*)carve(stateB);
  float*    cells      = (float*)carve(stateB);
  float*    hid_tmp    = (float*)carve(stateB);
  float*    hid_sorted = (float*)carve(stateB);
  float*    cells_tmp  = (float*)carve(stateB);
  _Float16* Acat       = (_Float16*)carve((size_t)MROW * KD * sizeof(_Float16));
  _Float16* Wcat       = (_Float16*)carve((size_t)GN * KD * sizeof(_Float16));
  float*    bcat       = (float*)carve(GN * sizeof(float));
  float*    hmean      = (float*)carve(BB * HH * sizeof(float));
  float*    cmean      = (float*)carve(BB * HH * sizeof(float));
  float*    newin      = (float*)carve(BB * INW * sizeof(float));
  float*    stdbuf     = (float*)carve(BB * HH * sizeof(float));
  float*    prob       = (float*)carve(MROW * sizeof(float));
  int*      sortidx    = (int*)carve(BB * KP * sizeof(int));
  int*      idx2       = (int*)carve(BB * KP * sizeof(int));

  // one-time setup
  k_zero<<<1024, 256, 0, stream>>>(hiddens, MROW * HH);
  k_zero<<<1024, 256, 0, stream>>>(cells,   MROW * HH);
  k_pack_w<<<(GN * KD + 255) / 256, 256, 0, stream>>>(W_ih, W_hh, b_ih, b_hh, Wcat, bcat);

  for (int t = 0; t < TT; ++t){
    k_state_mean<<<(BB * HH + 255) / 256, 256, 0, stream>>>(hiddens, cells, hmean, cmean);
    k_attn<<<BB, INW, 0, stream>>>(input_data, W_attn, b_attn, hmean, cmean, newin, out_iw, t);
    k_var_std<<<(BB * HH + 255) / 256, 256, 0, stream>>>(newin, hmean, W_var, b_var, stdbuf);
    k_pack_a<<<(MROW * KD + 255) / 256, 256, 0, stream>>>(newin, hiddens, Acat);
    k_lstm_wmma<<<128, 256, 0, stream>>>(Acat, Wcat, bcat, cells, stdbuf,
                                         hid_tmp, cells_tmp, t);
    k_fc_sort<<<BB, KP, 0, stream>>>(hid_tmp, W_fc, b_fc, sortidx);
    k_perm_sort<<<(MROW * HH + 255) / 256, 256, 0, stream>>>(hid_tmp, sortidx, hid_sorted);
    k_pdf<<<(MROW + 255) / 256, 256, 0, stream>>>(hid_sorted, newin, W_pdf, b_pdf, prob);
    k_resample<<<BB, KP, 0, stream>>>(prob, t, idx2);
    k_perm_resample<<<(MROW * HH + 255) / 256, 256, 0, stream>>>(hid_sorted, cells_tmp, idx2,
                                                                 hiddens, cells);
    k_enc_mean<<<HH, 256, 0, stream>>>(hiddens, out_enc, t);
  }
}
